// GraphAttentionLayer_58574763983363
// MI455X (gfx1250) — matrix operand order repaired and checked by reading, not compiled, not run on hardware
//
#include <hip/hip_runtime.h>

typedef __attribute__((ext_vector_type(16))) _Float16 v16h;
typedef __attribute__((ext_vector_type(8)))  float    v8f;
typedef __attribute__((ext_vector_type(4)))  float    v4f;

#define N_NODES 8192
#define F_IN    128
#define HEADS   4
#define F_OUT   64
#define OUT_DIM (HEADS * F_OUT)   // 256

// ---------------------------------------------------------------------------
// Kernel A: per-head projection feat = x @ W[h]  ->  featT (f16, [H][F_OUT][N])
// plus attention logits s[h][n] = feat . a_self, t[h][n] = feat . a_neigh.
// W[h] (32KB) staged in LDS; one thread per node per head; ~0.5 GFLOP total.
// ---------------------------------------------------------------------------
__global__ __launch_bounds__(256) void gat_project(
    const float* __restrict__ x, const float* __restrict__ W,
    const float* __restrict__ a_self, const float* __restrict__ a_neigh,
    _Float16* __restrict__ featT, float* __restrict__ s_out,
    float* __restrict__ t_out)
{
    __shared__ float Wl[F_IN * F_OUT];
    const int h   = blockIdx.y;
    const int tid = threadIdx.x;

    // stage W[h] into LDS (8192 floats)
    const v4f* Wg  = reinterpret_cast<const v4f*>(W + (size_t)h * F_IN * F_OUT);
    v4f*       Wl4 = reinterpret_cast<v4f*>(Wl);
    for (int i = tid; i < (F_IN * F_OUT) / 4; i += 256) Wl4[i] = Wg[i];
    __syncthreads();

    const int n = blockIdx.x * 256 + tid;
    const float* xr = x + (size_t)n * F_IN;

    v4f acc[F_OUT / 4];
#pragma unroll
    for (int j = 0; j < F_OUT / 4; ++j) acc[j] = (v4f){0.f, 0.f, 0.f, 0.f};

    for (int k = 0; k < F_IN; ++k) {
        const float xv = xr[k];
        const v4f* wr = reinterpret_cast<const v4f*>(Wl + k * F_OUT);
#pragma unroll
        for (int j = 0; j < F_OUT / 4; ++j) {
            v4f w = wr[j];
            acc[j].x = fmaf(xv, w.x, acc[j].x);
            acc[j].y = fmaf(xv, w.y, acc[j].y);
            acc[j].z = fmaf(xv, w.z, acc[j].z);
            acc[j].w = fmaf(xv, w.w, acc[j].w);
        }
    }

    const float* as = a_self  + h * F_OUT;
    const float* an = a_neigh + h * F_OUT;
    float sv = 0.f, tv = 0.f;
#pragma unroll
    for (int j = 0; j < F_OUT / 4; ++j) {
        sv = fmaf(acc[j].x, as[4*j+0], sv); sv = fmaf(acc[j].y, as[4*j+1], sv);
        sv = fmaf(acc[j].z, as[4*j+2], sv); sv = fmaf(acc[j].w, as[4*j+3], sv);
        tv = fmaf(acc[j].x, an[4*j+0], tv); tv = fmaf(acc[j].y, an[4*j+1], tv);
        tv = fmaf(acc[j].z, an[4*j+2], tv); tv = fmaf(acc[j].w, an[4*j+3], tv);
    }
    s_out[h * N_NODES + n] = sv;
    t_out[h * N_NODES + n] = tv;

    // transposed f16 features: featT[h][k][n] (coalesced across lanes on n)
    _Float16* ft = featT + (size_t)h * F_OUT * N_NODES + n;
#pragma unroll
    for (int j = 0; j < F_OUT / 4; ++j) {
        ft[(size_t)(4*j+0) * N_NODES] = (_Float16)acc[j].x;
        ft[(size_t)(4*j+1) * N_NODES] = (_Float16)acc[j].y;
        ft[(size_t)(4*j+2) * N_NODES] = (_Float16)acc[j].z;
        ft[(size_t)(4*j+3) * N_NODES] = (_Float16)acc[j].w;
    }
}

// ---------------------------------------------------------------------------
// Kernel B: fused masked-softmax attention + aggregation (flash-style).
// Block = 128 threads = 4 wave32, wave w handles head w; block handles a
// 16-row query tile. Inner loop streams 32-key tiles: builds the f16 P tile
// directly in WMMA A-fragment layout, loads feat B-fragments from featT, and
// accumulates 16x64 f32 output with v_wmma_f32_16x16x32_f16. Logits are
// bounded (|e| <~ 10) so exp() needs no max subtraction; masking is
// multiplicative (p = adj * exp(e)), matching softmax(e - 1e10*(1-adj)).
// Row-sum kept as a v4f partial vector -> 4 independent dependency chains
// (v_pk_add_f32) instead of one serial 16-add chain.
// ---------------------------------------------------------------------------
__global__ __launch_bounds__(128) void gat_attend(
    const float* __restrict__ adj, const float* __restrict__ s_in,
    const float* __restrict__ t_in, const _Float16* __restrict__ featT,
    const float* __restrict__ bias, float* __restrict__ out)
{
    const int h    = threadIdx.x >> 5;   // wave id == head
    const int lane = threadIdx.x & 31;
    const int nl   = lane & 15;          // A-row / C-col index
    const int grp  = lane >> 4;          // lane half-group
    const int base0 = grp ? 16 : 0;      // K offsets: contiguous k = 16*grp + s (matches B)
    const int base1 = grp ? 24 : 8;

    const int qbase = blockIdx.x * 16;

    const float   sval   = s_in[h * N_NODES + qbase + nl];
    const float*  adjRow = adj  + (size_t)(qbase + nl) * N_NODES;
    const float*  tRow   = t_in + (size_t)h * N_NODES;
    const _Float16* fT   = featT + (size_t)h * F_OUT * N_NODES;

    v8f cc0 = {}, cc1 = {}, cc2 = {}, cc3 = {};
    v4f rs = {};   // 4 independent row-sum partial chains

    for (int j0 = 0; j0 < N_NODES; j0 += 32) {
        // adjacency (streamed once -> non-temporal) and neighbor logits
        const v4f* a0p = reinterpret_cast<const v4f*>(adjRow + j0 + base0);
        const v4f* a1p = reinterpret_cast<const v4f*>(adjRow + j0 + base1);
        v4f A0 = __builtin_nontemporal_load(a0p);
        v4f A1 = __builtin_nontemporal_load(a0p + 1);
        v4f A2 = __builtin_nontemporal_load(a1p);
        v4f A3 = __builtin_nontemporal_load(a1p + 1);
        const v4f* t0p = reinterpret_cast<const v4f*>(tRow + j0 + base0);
        const v4f* t1p = reinterpret_cast<const v4f*>(tRow + j0 + base1);
        v4f T0 = t0p[0], T1 = t0p[1];
        v4f T2 = t1p[0], T3 = t1p[1];

        const float av[16] = {A0.x,A0.y,A0.z,A0.w, A1.x,A1.y,A1.z,A1.w,
                              A2.x,A2.y,A2.z,A2.w, A3.x,A3.y,A3.z,A3.w};
        const float tv[16] = {T0.x,T0.y,T0.z,T0.w, T1.x,T1.y,T1.z,T1.w,
                              T2.x,T2.y,T2.z,T2.w, T3.x,T3.y,T3.z,T3.w};

        float p[16];
#pragma unroll
        for (int i = 0; i < 16; ++i) {
            float e = sval + tv[i];
            e = fmaxf(e, 0.2f * e);              // LeakyReLU(0.2)
            p[i] = av[i] * __expf(e);            // adjacency mask (binary)
        }

        // 4 independent accumulation chains (pairs fuse into v_pk_add_f32)
#pragma unroll
        for (int q = 0; q < 4; ++q) {
            v4f chunk = {p[4*q+0], p[4*q+1], p[4*q+2], p[4*q+3]};
            rs += chunk;
        }

        v16h af;
#pragma unroll
        for (int i = 0; i < 16; ++i) af[i] = (_Float16)p[i];

        // B fragments: featT rows are contiguous in K -> single 32B load each
        const size_t bofs = (size_t)j0 + (size_t)grp * 16;
        v16h b0 = *reinterpret_cast<const v16h*>(fT + (size_t)(0*16 + nl) * N_NODES + bofs);
        v16h b1 = *reinterpret_cast<const v16h*>(fT + (size_t)(1*16 + nl) * N_NODES + bofs);
        v16h b2 = *reinterpret_cast<const v16h*>(fT + (size_t)(2*16 + nl) * N_NODES + bofs);
        v16h b3 = *reinterpret_cast<const v16h*>(fT + (size_t)(3*16 + nl) * N_NODES + bofs);

        cc0 = __builtin_amdgcn_wmma_f32_16x16x32_f16(false, af, false, b0, (short)0, cc0, false, false);
        cc1 = __builtin_amdgcn_wmma_f32_16x16x32_f16(false, af, false, b1, (short)0, cc1, false, false);
        cc2 = __builtin_amdgcn_wmma_f32_16x16x32_f16(false, af, false, b2, (short)0, cc2, false, false);
        cc3 = __builtin_amdgcn_wmma_f32_16x16x32_f16(false, af, false, b3, (short)0, cc3, false, false);
    }

    // reduce the 4 partial chains, then combine the two half-row partials
    // (lanes l and l+16 share row l%16)
    float rowsum = (rs.x + rs.y) + (rs.z + rs.w);
    float tot = rowsum + __shfl_xor(rowsum, 16);

    // per-C-row reciprocal sums: row of VGPR r is r + 8*grp; lane m holds row m
    float inv[8];
#pragma unroll
    for (int r = 0; r < 8; ++r)
        inv[r] = 1.0f / __shfl(tot, r + 8 * grp);

    const float bv0 = bias[h * F_OUT + 0*16 + nl];
    const float bv1 = bias[h * F_OUT + 1*16 + nl];
    const float bv2 = bias[h * F_OUT + 2*16 + nl];
    const float bv3 = bias[h * F_OUT + 3*16 + nl];

#pragma unroll
    for (int r = 0; r < 8; ++r) {
        const int row = r + 8 * grp;
        float* orow = out + (size_t)(qbase + row) * OUT_DIM + h * F_OUT + nl;
        orow[0*16] = fmaxf(fmaf(cc0[r], inv[r], bv0), 0.f);
        orow[1*16] = fmaxf(fmaf(cc1[r], inv[r], bv1), 0.f);
        orow[2*16] = fmaxf(fmaf(cc2[r], inv[r], bv2), 0.f);
        orow[3*16] = fmaxf(fmaf(cc3[r], inv[r], bv3), 0.f);
    }
}

// ---------------------------------------------------------------------------
extern "C" void kernel_launch(void* const* d_in, const int* in_sizes, int n_in,
                              void* d_out, int out_size, void* d_ws, size_t ws_size,
                              hipStream_t stream)
{
    const float* x       = (const float*)d_in[0];   // [N, F]
    const float* adj     = (const float*)d_in[1];   // [N, N]
    const float* W       = (const float*)d_in[2];   // [H, F, F_]
    const float* a_self  = (const float*)d_in[3];   // [H, F_]
    const float* a_neigh = (const float*)d_in[4];   // [H, F_]
    const float* bias    = (const float*)d_in[5];   // [H, F_]
    float*       out     = (float*)d_out;           // [N, H*F_]

    // workspace: featT f16 [H][F_][N] (4MB) | s f32 [H][N] | t f32 [H][N]
    _Float16* featT = (_Float16*)d_ws;
    float* s = (float*)((char*)d_ws + (size_t)HEADS * F_OUT * N_NODES * sizeof(_Float16));
    float* t = s + (size_t)HEADS * N_NODES;

    dim3 gA(N_NODES / 256, HEADS);
    gat_project<<<gA, 256, 0, stream>>>(x, W, a_self, a_neigh, featT, s, t);

    gat_attend<<<N_NODES / 16, 128, 0, stream>>>(adj, s, t, featT, bias, out);
}
